// IBNCell_57148834840927
// MI455X (gfx1250) — matrix operand order repaired and checked by reading, not compiled, not run on hardware
//
#include <hip/hip_runtime.h>

// ---------------------------------------------------------------------------
// IBN cell for MI455X (gfx1250). Heavy GEMMs run on v_wmma_f32_16x16x32_bf16
// using bf16x3 split-precision (hi*hi + hi*lo + lo*hi) for near-fp32 accuracy.
// bf16 split happens ONCE at LDS staging; fragments load as ds_load_b128.
// Fast unguarded K-loop for K % 32 == 0 (all heavy GEMMs); guarded tail path
// only for the tiny K=16 embedding GEMM. Workspace requirement: ~118 MB.
// ---------------------------------------------------------------------------

#define Bb   16
#define Nn   1024
#define Cc   64
#define CIn  16
#define NNx  (1024 * 1024)
#define BCx  (64 * 1024)

typedef __attribute__((ext_vector_type(16))) __bf16       v16bf;
typedef __attribute__((ext_vector_type(8)))  float        v8f;
typedef __attribute__((ext_vector_type(4)))  unsigned int u32x4;

union F8     { v8f   v; float a[8]; };
union FragBF { v16bf v; u32x4 q[2]; };

__device__ inline float lrelu(float v) { return v > 0.f ? v : 0.01f * v; }

// pack two consecutive-k floats into one dword of bf16-hi and one of bf16-lo
__device__ inline void bf_split_pack(float f0, float f1, unsigned& hi, unsigned& lo) {
  unsigned u0 = __float_as_uint(f0), u1 = __float_as_uint(f1);
  hi = (u0 >> 16) | (u1 & 0xFFFF0000u);
  float l0 = f0 - __uint_as_float(u0 & 0xFFFF0000u);
  float l1 = f1 - __uint_as_float(u1 & 0xFFFF0000u);
  lo = (__float_as_uint(l0) >> 16) | (__float_as_uint(l1) & 0xFFFF0000u);
}

// ---------------------------------------------------------------------------
// Generic strided batched GEMM: C[m,n] (+)= epi( sum_k A[m,k]*B[k,n] )
// epi: 0 none | 1 +biasM | 2 +biasN | 3 +biasM,lrelu | 4 *1/Z[m],relu
//      5 gl-weights: exp( exp(-max(sq[m]+sq[n]-2acc,0)/128) + (m==n) )
// ---------------------------------------------------------------------------
struct GemmP {
  const float* A;  long long aRS, aCS, aB;
  const float* Bm; long long bRS, bCS, bB;
  float*       C;  long long cLD, cB;
  int M, N, K;
  int epi, accum;
  const float* v1; long long v1B;
};

#define BM 64
#define BN 64
#define BK 32
#define LDK 40   // padded row length (80 bytes, 16B-aligned, bank-skewed)

__device__ inline float apply_epi(const GemmP& p, float acc, const float* vp,
                                  int gm, int gn) {
  switch (p.epi) {
    case 1: return acc + vp[gm];
    case 2: return acc + vp[gn];
    case 3: return lrelu(acc + vp[gm]);
    case 4: { float v = acc / vp[gm]; return v > 0.f ? v : 0.f; }
    case 5: {
      float d = vp[gm] + vp[gn] - 2.f * acc;
      if (d < 0.f) d = 0.f;
      float z = expf(-d * (1.f / 128.f)) + (gm == gn ? 1.f : 0.f);
      return expf(z);   // softmax numerator; z <= 2 so no max-shift needed
    }
    default: return acc;
  }
}

// fragment gather (2x ds_load_b128 per plane) + 6x WMMA bf16x3
__device__ inline void mma_step(const __bf16 (*Ah)[LDK], const __bf16 (*Al)[LDK],
                                const __bf16 (*Bh)[LDK], const __bf16 (*Bl)[LDK],
                                int wr, int am, int akb, int wc, int bn, int bkb,
                                F8& a0, F8& a1) {
  FragBF ah, al, bh0, bl0, bh1, bl1;
  ah.q[0]  = *(const u32x4*)&Ah[wr + am][akb];
  ah.q[1]  = *(const u32x4*)&Ah[wr + am][akb + 8];
  al.q[0]  = *(const u32x4*)&Al[wr + am][akb];
  al.q[1]  = *(const u32x4*)&Al[wr + am][akb + 8];
  bh0.q[0] = *(const u32x4*)&Bh[wc + bn][bkb];
  bh0.q[1] = *(const u32x4*)&Bh[wc + bn][bkb + 8];
  bl0.q[0] = *(const u32x4*)&Bl[wc + bn][bkb];
  bl0.q[1] = *(const u32x4*)&Bl[wc + bn][bkb + 8];
  bh1.q[0] = *(const u32x4*)&Bh[wc + 16 + bn][bkb];
  bh1.q[1] = *(const u32x4*)&Bh[wc + 16 + bn][bkb + 8];
  bl1.q[0] = *(const u32x4*)&Bl[wc + 16 + bn][bkb];
  bl1.q[1] = *(const u32x4*)&Bl[wc + 16 + bn][bkb + 8];
  // bf16x3: hi*hi + hi*lo + lo*hi
  a0.v = __builtin_amdgcn_wmma_f32_16x16x32_bf16(false, ah.v, false, bh0.v, (short)0, a0.v, false, false);
  a0.v = __builtin_amdgcn_wmma_f32_16x16x32_bf16(false, ah.v, false, bl0.v, (short)0, a0.v, false, false);
  a0.v = __builtin_amdgcn_wmma_f32_16x16x32_bf16(false, al.v, false, bh0.v, (short)0, a0.v, false, false);
  a1.v = __builtin_amdgcn_wmma_f32_16x16x32_bf16(false, ah.v, false, bh1.v, (short)0, a1.v, false, false);
  a1.v = __builtin_amdgcn_wmma_f32_16x16x32_bf16(false, ah.v, false, bl1.v, (short)0, a1.v, false, false);
  a1.v = __builtin_amdgcn_wmma_f32_16x16x32_bf16(false, al.v, false, bh1.v, (short)0, a1.v, false, false);
}

__global__ __launch_bounds__(256) void gemm_bf16x3(GemmP p) {
  // bf16 hi/lo planes; A row-major [m][k], B transposed [n][k], K-contiguous
  __shared__ __bf16 Ah[BM][LDK], Al[BM][LDK];
  __shared__ __bf16 Bh[BN][LDK], Bl[BN][LDK];

  const int bz = blockIdx.z;
  const float* A = p.A  + (long long)bz * p.aB;
  const float* B = p.Bm + (long long)bz * p.bB;
  float*       C = p.C  + (long long)bz * p.cB;

  const int m0 = blockIdx.y * BM, n0 = blockIdx.x * BN;
  const int tid  = threadIdx.x;
  const int lane = tid & 31;
  const int wave = tid >> 5;
  const int wr   = (wave & 3) * 16;   // wave row block 0..48
  const int wc   = (wave >> 2) * 32;  // wave col block 0/32

  // 16-bit WMMA fragment coordinates used here (contiguous k order, same for A and B)
  const int am  = lane & 15;          // A: M within tile
  const int akb = (lane >> 4) * 16;   // A: K base (0 or 16); elem e -> kb+e
  const int bn  = lane & 15;          // B: N within half-tile
  const int bkb = (lane >> 4) * 16;   // B: K base (0 or 16); elem e -> kb+e

  // staging coordinates: 1024 k-pairs per tile, 4 per thread
  const int sm = tid >> 2;            // row (A: m, B: n), 0..63
  const int sk = (tid & 3) * 8;       // k base for this thread's 4 pairs

  // per-thread staging base pointers (k advances by aCS/bRS)
  const float* aS = A + (long long)(m0 + sm) * p.aRS + (long long)sk * p.aCS;
  const float* bS = B + (long long)(n0 + sm) * p.bCS + (long long)sk * p.bRS;

  F8 a0, a1; a0.v = (v8f){}; a1.v = (v8f){};

  if ((p.K & (BK - 1)) == 0) {
    // ---- fast path: no K-tail, unguarded loads, unconditional prefetch ----
    for (int k0 = 0; k0 < p.K; k0 += BK) {
#pragma unroll
      for (int kp = 0; kp < 4; ++kp) {
        int k = sk + kp * 2;
        const float* ap = aS + (long long)(k0 + kp * 2) * p.aCS;
        const float* bp = bS + (long long)(k0 + kp * 2) * p.bRS;
        unsigned hi, lo;
        bf_split_pack(ap[0], ap[p.aCS], hi, lo);
        __builtin_prefetch(ap + BK * p.aCS, 0, 1);   // speculative, non-faulting
        *(unsigned*)&Ah[sm][k] = hi;
        *(unsigned*)&Al[sm][k] = lo;
        bf_split_pack(bp[0], bp[p.bRS], hi, lo);
        __builtin_prefetch(bp + BK * p.bRS, 0, 1);
        *(unsigned*)&Bh[sm][k] = hi;
        *(unsigned*)&Bl[sm][k] = lo;
      }
      __syncthreads();
      mma_step(Ah, Al, Bh, Bl, wr, am, akb, wc, bn, bkb, a0, a1);
      __syncthreads();
    }
  } else {
    // ---- tail path: guarded loads (only the K=16 embedding GEMM) ----
    for (int k0 = 0; k0 < p.K; k0 += BK) {
#pragma unroll
      for (int kp = 0; kp < 4; ++kp) {
        int k = sk + kp * 2;
        int gk0 = k0 + k, gk1 = gk0 + 1;
        const float* ap = aS + (long long)(k0 + kp * 2) * p.aCS;
        const float* bp = bS + (long long)(k0 + kp * 2) * p.bRS;
        float fa0 = (gk0 < p.K) ? ap[0] : 0.f;
        float fa1 = (gk1 < p.K) ? ap[p.aCS] : 0.f;
        float fb0 = (gk0 < p.K) ? bp[0] : 0.f;
        float fb1 = (gk1 < p.K) ? bp[p.bRS] : 0.f;
        unsigned hi, lo;
        bf_split_pack(fa0, fa1, hi, lo);
        *(unsigned*)&Ah[sm][k] = hi;
        *(unsigned*)&Al[sm][k] = lo;
        bf_split_pack(fb0, fb1, hi, lo);
        *(unsigned*)&Bh[sm][k] = hi;
        *(unsigned*)&Bl[sm][k] = lo;
      }
      __syncthreads();
      mma_step(Ah, Al, Bh, Bl, wr, am, akb, wc, bn, bkb, a0, a1);
      __syncthreads();
    }
  }

  const float* vp = p.v1 ? (p.v1 + (long long)bz * p.v1B) : nullptr;
  const int mhalf = (lane >> 4) * 8;   // C/D layout: lanes 16-31 hold M=8..15
#pragma unroll
  for (int r = 0; r < 8; ++r) {
    int gm = m0 + wr + mhalf + r;
    long long ro = (long long)gm * p.cLD;
    int g0 = n0 + wc + bn, g1 = g0 + 16;
    float v0 = apply_epi(p, a0.a[r], vp, gm, g0);
    float v1 = apply_epi(p, a1.a[r], vp, gm, g1);
    if (p.accum) { v0 += C[ro + g0]; v1 += C[ro + g1]; }
    C[ro + g0] = v0;
    C[ro + g1] = v1;
  }
}

// ---------------------------------------------------------------------------
// Small fused kernels
// ---------------------------------------------------------------------------

// e1/e2 = h . att_a[:C] / att_a[C:]
__global__ __launch_bounds__(256) void k_e12(const float* H, const float* a,
                                             float* e1, float* e2) {
  int idx = blockIdx.x * 256 + threadIdx.x;       // (b,n)
  const float* hr = H + (long long)idx * Cc;
  float s1 = 0.f, s2 = 0.f;
#pragma unroll 8
  for (int c = 0; c < Cc; ++c) { float h = hr[c]; s1 += h * a[c]; s2 += h * a[Cc + c]; }
  e1[idx] = s1; e2[idx] = s2;
}

__global__ __launch_bounds__(256) void k_maxe2(const float* e2, float* me2) {
  __shared__ float red[256];
  int b = blockIdx.x;
  float m = -3.0e38f;
  for (int t = 0; t < 4; ++t) { float v = e2[b * Nn + threadIdx.x + t * 256]; m = v > m ? v : m; }
  red[threadIdx.x] = m; __syncthreads();
  for (int st = 128; st > 0; st >>= 1) {
    if (threadIdx.x < st) { float o = red[threadIdx.x + st]; if (o > red[threadIdx.x]) red[threadIdx.x] = o; }
    __syncthreads();
  }
  if (threadIdx.x == 0) me2[b] = red[0];
}

// attention row: w = exp(lrelu(e1_i+e2_j) - lrelu(e1_i+max e2)); Z = sum_j w
__global__ __launch_bounds__(256) void k_attrow(const float* e1, const float* e2,
                                                const float* me2, float* P, float* Z) {
  __shared__ float red[256];
  int blk = blockIdx.x, b = blk >> 10, i = blk & 1023;
  float a = e1[b * Nn + i];
  float mrow = lrelu(a + me2[b]);
  float* row = P + (long long)b * NNx + (long long)i * Nn;
  float s = 0.f;
#pragma unroll
  for (int t = 0; t < 4; ++t) {
    int j = threadIdx.x + t * 256;
    float w = expf(lrelu(a + e2[b * Nn + j]) - mrow);
    row[j] = w; s += w;
  }
  red[threadIdx.x] = s; __syncthreads();
  for (int st = 128; st > 0; st >>= 1) {
    if (threadIdx.x < st) red[threadIdx.x] += red[threadIdx.x + st];
    __syncthreads();
  }
  if (threadIdx.x == 0) Z[b * Nn + i] = red[0];
}

__global__ __launch_bounds__(256) void k_sq(const float* MU, float* SQ) {
  int idx = blockIdx.x * 256 + threadIdx.x;       // (b,n)
  const float* r = MU + (long long)idx * Cc;
  float s = 0.f;
#pragma unroll 8
  for (int c = 0; c < Cc; ++c) s += r[c] * r[c];
  SQ[idx] = s;
}

// row-normalize gl weights in place (softmax denominator)
__global__ __launch_bounds__(256) void k_glrow(float* P) {
  __shared__ float red[256];
  int blk = blockIdx.x, b = blk >> 10, i = blk & 1023;
  float* row = P + (long long)b * NNx + (long long)i * Nn;
  float v[4]; float s = 0.f;
#pragma unroll
  for (int t = 0; t < 4; ++t) { v[t] = row[threadIdx.x + t * 256]; s += v[t]; }
  red[threadIdx.x] = s; __syncthreads();
  for (int st = 128; st > 0; st >>= 1) {
    if (threadIdx.x < st) red[threadIdx.x] += red[threadIdx.x + st];
    __syncthreads();
  }
  float inv = 1.f / red[0];
#pragma unroll
  for (int t = 0; t < 4; ++t) row[threadIdx.x + t * 256] = v[t] * inv;
}

// laplacian degree: d_i = (sum_j (g[i,j]) + 1)^(-1/2)   (g + I)
__global__ __launch_bounds__(256) void k_deg(const float* G, float* DG) {
  int idx = blockIdx.x * 256 + threadIdx.x;       // (g,i), 2048 total
  int g = idx >> 10, i = idx & 1023;
  const float* row = G + (long long)g * NNx + (long long)i * Nn;
  float s = 1.f;
  for (int j = 0; j < Nn; ++j) s += row[j];
  DG[idx] = rsqrtf(s);
}

// out[i,j] = (g[j,i] + (i==j)) * d_i * d_j
__global__ __launch_bounds__(256) void k_lap(const float* G, const float* DG,
                                             float* G1, float* G2) {
  long long idx = (long long)blockIdx.x * 256 + threadIdx.x;  // 2*N*N
  int g = (int)(idx >> 20), i = (int)((idx >> 10) & 1023), j = (int)(idx & 1023);
  float m = G[(long long)g * NNx + (long long)j * Nn + i] + (i == j ? 1.f : 0.f);
  float v = m * DG[g * Nn + i] * DG[g * Nn + j];
  (g == 0 ? G1 : G2)[(long long)i * Nn + j] = v;
}

// LayerNorm over [C,N] per batch, optional exact GELU
__global__ __launch_bounds__(1024) void k_ln(const float* in, float* out,
                                             const float* w, const float* b,
                                             int do_gelu) {
  __shared__ float s1[1024], s2[1024];
  int bz = blockIdx.x;
  const float* z = in + (long long)bz * BCx;
  float sum = 0.f, sq = 0.f;
  for (int i = threadIdx.x; i < BCx; i += 1024) { float v = z[i]; sum += v; sq += v * v; }
  s1[threadIdx.x] = sum; s2[threadIdx.x] = sq; __syncthreads();
  for (int st = 512; st > 0; st >>= 1) {
    if (threadIdx.x < st) { s1[threadIdx.x] += s1[threadIdx.x + st]; s2[threadIdx.x] += s2[threadIdx.x + st]; }
    __syncthreads();
  }
  float mean = s1[0] * (1.f / BCx);
  float var  = s2[0] * (1.f / BCx) - mean * mean;
  float rstd = rsqrtf(var + 1e-5f);
  for (int i = threadIdx.x; i < BCx; i += 1024) {
    float v = (z[i] - mean) * rstd * w[i] + b[i];
    if (do_gelu) v = 0.5f * v * (1.f + erff(v * 0.70710678118f));
    out[(long long)bz * BCx + i] = v;
  }
}

// ct_new = ft*ct + xn - ft*xn ; ht = ft*elu(ct_new) + xu - ft*xu
__global__ __launch_bounds__(256) void k_combine(const float* FT, const float* XN,
                                                 const float* MU, const float* CT,
                                                 float* out) {
  long long idx = (long long)blockIdx.x * 256 + threadIdx.x;  // b*65536 + c*1024 + n
  int b = (int)(idx >> 16), rem = (int)(idx & 65535);
  int c = rem >> 10, n = rem & 1023;
  float xu = MU[(long long)b * BCx + (long long)n * Cc + c];  // transpose gather
  float ft = FT[idx], xn = XN[idx], ct = CT[idx];
  float ctn = ft * ct + xn - ft * xn;
  float el  = ctn > 0.f ? ctn : (expf(ctn) - 1.f);
  float ht  = ft * el + xu - ft * xu;
  out[idx] = ht;
  out[(long long)Bb * BCx + idx] = ctn;
}

// ---------------------------------------------------------------------------
// Host side
// ---------------------------------------------------------------------------
static inline void gemm(hipStream_t s,
                        const float* A, long long aRS, long long aCS, long long aB,
                        const float* Bm, long long bRS, long long bCS, long long bB,
                        float* C, long long cLD, long long cB,
                        int M, int N, int K, int epi, int accum,
                        const float* v1, long long v1B, int batch) {
  GemmP p{A, aRS, aCS, aB, Bm, bRS, bCS, bB, C, cLD, cB, M, N, K, epi, accum, v1, v1B};
  dim3 g((unsigned)(N / 64), (unsigned)(M / 64), (unsigned)batch);
  hipLaunchKernelGGL(gemm_bf16x3, g, dim3(256), 0, s, p);
}

extern "C" void kernel_launch(void* const* d_in, const int* in_sizes, int n_in,
                              void* d_out, int out_size, void* d_ws, size_t ws_size,
                              hipStream_t stream) {
  (void)in_sizes; (void)n_in; (void)out_size; (void)ws_size;
  const float* x      = (const float*)d_in[0];
  const float* ct     = (const float*)d_in[1];
  const float* graph  = (const float*)d_in[2];
  const float* emb_w  = (const float*)d_in[3];
  const float* emb_b  = (const float*)d_in[4];
  const float* emb2_w = (const float*)d_in[5];
  const float* emb2_b = (const float*)d_in[6];
  const float* att_W  = (const float*)d_in[7];
  const float* att_a  = (const float*)d_in[8];
  // d_in[9] = att_GL: dead code (softmax output is strictly positive -> mask always true)
  const float* uai_w  = (const float*)d_in[10];
  const float* uai_b  = (const float*)d_in[11];
  const float* lin1_w = (const float*)d_in[12];
  const float* lin2_w = (const float*)d_in[13];
  const float* lin2_b = (const float*)d_in[14];
  const float* ln_w   = (const float*)d_in[15];
  const float* ln_b   = (const float*)d_in[16];
  float* out = (float*)d_out;

  const long long SL = 1048576;  // one [B,64,1024] fp32 buffer (in floats)
  float* W   = (float*)d_ws;
  float* T0  = W;            // t0 -> later atth
  float* H0  = W + 1 * SL;   // h0 -> later l1x
  float* H   = W + 2 * SL;   // h  -> later l2x
  float* MU  = W + 3 * SL;   // x_uai as [B,N,C]
  float* TA  = W + 4 * SL;   // x_new pre-LN accum
  float* TB  = W + 5 * SL;   // ft pre-LN accum
  float* SC1 = W + 6 * SL;
  float* SC2 = W + 7 * SL;
  float* XN  = W + 8 * SL;
  float* FT  = W + 9 * SL;
  float* G1  = W + 10 * SL;
  float* G2  = W + 11 * SL;
  float* P   = W + 12 * SL;  // [B,N,N]: att weights, then gl weights (16*SL)
  float* SM  = W + 28 * SL;
  float* E1  = SM;             // B*N
  float* E2  = SM + 16384;     // B*N
  float* Zr  = SM + 32768;     // B*N
  float* SQ  = SM + 49152;     // B*N
  float* ME2 = SM + 65536;     // B
  float* DG  = SM + 65536 + 64; // 2*N

  // 1) t0 = lrelu(emb_w @ x + emb_b)              [B,64,1024]
  gemm(stream, emb_w, 16, 1, 0,  x, 1024, 1, (long long)CIn * Nn,
       T0, 1024, BCx, 64, 1024, 16, 3, 0, emb_b, 0, Bb);
  // 2) h0 = t0 @ emb2_w^T + emb2_b                [B,64,1024]
  gemm(stream, T0, 1024, 1, BCx,  emb2_w, 1, 1024, 0,
       H0, 1024, BCx, 64, 1024, 1024, 2, 0, emb2_b, 0, Bb);
  // 3) h = inp @ att_W                            [B,1024,64]
  gemm(stream, H0, 1, 1024, BCx,  att_W, 64, 1, 0,
       H, 64, BCx, 1024, 64, 64, 0, 0, nullptr, 0, Bb);
  // 4-6) attention scores -> P (unnormalized), Z row sums
  hipLaunchKernelGGL(k_e12, dim3(64), dim3(256), 0, stream, H, att_a, E1, E2);
  hipLaunchKernelGGL(k_maxe2, dim3(16), dim3(256), 0, stream, E2, ME2);
  hipLaunchKernelGGL(k_attrow, dim3(16384), dim3(256), 0, stream, E1, E2, ME2, P, Zr);
  // 7) atth = relu((P @ h)/Z)  -> reuse T0        [B,1024,64]
  gemm(stream, P, 1024, 1, (long long)NNx,  H, 64, 1, BCx,
       T0, 64, BCx, 1024, 64, 1024, 4, 0, Zr, 1024, Bb);
  // 8) mu = atth @ uai_w^T + uai_b                [B,1024,64]
  gemm(stream, T0, 64, 1, BCx,  uai_w, 1, 64, 0,
       MU, 64, BCx, 1024, 64, 64, 2, 0, uai_b, 0, Bb);
  // 9-11) gl weights: exp(exp(-d2/128)+I) via mu@mu^T epilogue, then row-norm
  hipLaunchKernelGGL(k_sq, dim3(64), dim3(256), 0, stream, MU, SQ);
  gemm(stream, MU, 64, 1, BCx,  MU, 1, 64, BCx,
       P, 1024, (long long)NNx, 1024, 1024, 64, 5, 0, SQ, 1024, Bb);
  hipLaunchKernelGGL(k_glrow, dim3(16384), dim3(256), 0, stream, P);
  // 12-13) laplacians g1,g2
  hipLaunchKernelGGL(k_deg, dim3(8), dim3(256), 0, stream, graph, DG);
  hipLaunchKernelGGL(k_lap, dim3(8192), dim3(256), 0, stream, graph, DG, G1, G2);
  // l1 chain -> TA
  gemm(stream, lin1_w, 64, 1, 0,  MU, 1, 64, BCx,
       H0, 1024, BCx, 64, 1024, 64, 0, 0, nullptr, 0, Bb);             // l1x
  gemm(stream, H0, 1024, 1, BCx,  G1, 1024, 1, 0,
       SC1, 1024, BCx, 64, 1024, 1024, 0, 0, nullptr, 0, Bb);          // l1x@g1
  gemm(stream, lin1_w, 64, 1, 0,  SC1, 1024, 1, BCx,
       SC2, 1024, BCx, 64, 1024, 64, 0, 0, nullptr, 0, Bb);            // l1(l1x@g1)
  gemm(stream, SC2, 1024, 1, BCx,  G2, 1024, 1, 0,
       TA, 1024, BCx, 64, 1024, 1024, 0, 0, nullptr, 0, Bb);           // @g2
  gemm(stream, H0, 1024, 1, BCx,  P, 1024, 1, (long long)NNx,
       TA, 1024, BCx, 64, 1024, 1024, 0, 1, nullptr, 0, Bb);           // += l1x@gl
  // l2 chain -> TB
  gemm(stream, lin2_w, 64, 1, 0,  MU, 1, 64, BCx,
       H, 1024, BCx, 64, 1024, 64, 1, 0, lin2_b, 0, Bb);               // l2x
  gemm(stream, H, 1024, 1, BCx,  G1, 1024, 1, 0,
       SC1, 1024, BCx, 64, 1024, 1024, 0, 0, nullptr, 0, Bb);
  gemm(stream, lin2_w, 64, 1, 0,  SC1, 1024, 1, BCx,
       SC2, 1024, BCx, 64, 1024, 64, 1, 0, lin2_b, 0, Bb);
  gemm(stream, SC2, 1024, 1, BCx,  G2, 1024, 1, 0,
       TB, 1024, BCx, 64, 1024, 1024, 0, 0, nullptr, 0, Bb);
  gemm(stream, H, 1024, 1, BCx,  P, 1024, 1, (long long)NNx,
       TB, 1024, BCx, 64, 1024, 1024, 0, 1, nullptr, 0, Bb);
  // LN / LN+GELU, then combine
  hipLaunchKernelGGL(k_ln, dim3(16), dim3(1024), 0, stream, TA, XN, ln_w, ln_b, 0);
  hipLaunchKernelGGL(k_ln, dim3(16), dim3(1024), 0, stream, TB, FT, ln_w, ln_b, 1);
  hipLaunchKernelGGL(k_combine, dim3(4096), dim3(256), 0, stream, FT, XN, MU, ct, out);
}
